// LengthRegulator_46634754900528
// MI455X (gfx1250) — matrix operand order repaired and checked by reading, not compiled, not run on hardware
//
#include <hip/hip_runtime.h>
#include <hip/hip_bf16.h>

typedef __attribute__((ext_vector_type(16))) __bf16 v16bf;
typedef __attribute__((ext_vector_type(8)))  __bf16 v8bf;
typedef __attribute__((ext_vector_type(8)))  float  v8f;
typedef __attribute__((ext_vector_type(4)))  float  f4v;

#define T_ENC 512
#define T_DEC 2048
#define DDIM  256

#define BM 128
#define BN 256
#define BK 32
#define NK (T_ENC / BK)
#define LDA 40   // padded row stride (halves): 80B rows -> conflict-free b128 LDS reads
#define LDB 40

#define CAT16(a,b) __builtin_shufflevector((a),(b),0,1,2,3,4,5,6,7,8,9,10,11,12,13,14,15)

__global__ __launch_bounds__(256)
void bmm_wmma_bf16x3(const float* __restrict__ x,
                     const float* __restrict__ path,
                     float* __restrict__ out)
{
    // Double-buffered panels: A hi/lo 40 KB + B^T hi/lo 80 KB = 120 KB (WGP has 320 KB)
    __shared__ __bf16 AhiS[2][BM * LDA];
    __shared__ __bf16 AloS[2][BM * LDA];
    __shared__ __bf16 BhiS[2][BN * LDB];   // transposed: [n][k]
    __shared__ __bf16 BloS[2][BN * LDB];

    const int b  = blockIdx.z;
    const int m0 = blockIdx.y * BM;
    const int n0 = blockIdx.x * BN;

    const float* xb = x    + (size_t)b * DDIM  * T_ENC;
    const float* pb = path + (size_t)b * T_ENC * T_DEC;
    float*       ob = out  + (size_t)b * DDIM  * T_DEC;

    const int t     = threadIdx.x;
    const int lane  = t & 31;
    const int wave  = t >> 5;
    const int waveM = (wave & 1) * 64;   // 2 waves tile M
    const int waveN = (wave >> 1) * 64;  // 4 waves tile N
    const int hg    = lane >> 4;         // half-group (lanes 0-15 / 16-31)
    const int lm    = lane & 15;

    // per-thread staging coordinates (contiguous 8-element K-runs -> b128 LDS stores)
    const int ar  = t >> 2;         // A: row (0..63); second pass adds 64
    const int ac  = (t & 3) * 8;    // A: K chunk base (0,8,16,24)
    const int bkk = (t >> 6) * 8;   // B: contiguous K-run base (0,8,16,24)
    const int bc  = (t & 63) * 4;   // B: column group base (0..252)

    f4v aReg[4];   // A: 2 row-passes x 2 float4 (8 contiguous K each)
    f4v bReg[8];   // B: 8 K-rows x float4 (4 columns)

    auto load_regs = [&](int k0) {
#pragma unroll
        for (int i = 0; i < 2; ++i) {
            const float* src = &xb[(size_t)(m0 + ar + 64 * i) * T_ENC + k0 + ac];
            aReg[2 * i + 0] = *(const f4v*)(src + 0);
            aReg[2 * i + 1] = *(const f4v*)(src + 4);
        }
#pragma unroll
        for (int i = 0; i < 8; ++i)
            bReg[i] = *(const f4v*)&pb[(size_t)(k0 + bkk + i) * T_DEC + n0 + bc];
    };

    auto store_lds = [&](int buf) {
        __bf16* Ahi = AhiS[buf];  __bf16* Alo = AloS[buf];
        __bf16* Bhi = BhiS[buf];  __bf16* Blo = BloS[buf];
        // A: one b128 hi + one b128 lo per row-pass
#pragma unroll
        for (int i = 0; i < 2; ++i) {
            v8bf hv, lv;
#pragma unroll
            for (int e = 0; e < 8; ++e) {
                const float  f = aReg[2 * i + (e >> 2)][e & 3];
                const __bf16 h = (__bf16)f;
                hv[e] = h;
                lv[e] = (__bf16)(f - (float)h);
            }
            *(v8bf*)&Ahi[(ar + 64 * i) * LDA + ac] = hv;
            *(v8bf*)&Alo[(ar + 64 * i) * LDA + ac] = lv;
        }
        // B: per column, pack 8 contiguous K values -> one b128 hi + one b128 lo
#pragma unroll
        for (int j = 0; j < 4; ++j) {
            v8bf hv, lv;
#pragma unroll
            for (int i = 0; i < 8; ++i) {
                const float  f = bReg[i][j];
                const __bf16 h = (__bf16)f;
                hv[i] = h;
                lv[i] = (__bf16)(f - (float)h);
            }
            *(v8bf*)&Bhi[(bc + j) * LDB + bkk] = hv;   // transposed [n][k]
            *(v8bf*)&Blo[(bc + j) * LDB + bkk] = lv;
        }
    };

    v8f acc[4][4];
    const v8f vzero = {0.f, 0.f, 0.f, 0.f, 0.f, 0.f, 0.f, 0.f};
#pragma unroll
    for (int mi = 0; mi < 4; ++mi)
#pragma unroll
        for (int ni = 0; ni < 4; ++ni)
            acc[mi][ni] = vzero;

    // ---- prologue: stage panel 0 ----
    load_regs(0);
    store_lds(0);
    __syncthreads();

    for (int it = 0; it < NK; ++it) {
        const int  cur  = it & 1;
        const bool more = (it + 1) < NK;

        // issue next panel's global loads early; they fly under the WMMA burst
        if (more) load_regs((it + 1) * BK);

        const __bf16* Ahi = AhiS[cur];  const __bf16* Alo = AloS[cur];
        const __bf16* Bhi = BhiS[cur];  const __bf16* Blo = BloS[cur];

        // ---- B fragments (reused by all 4 mi) ----
        v16bf bh[4], bl[4];
#pragma unroll
        for (int ni = 0; ni < 4; ++ni) {
            const int col = waveN + ni * 16 + lm;   // B: lane holds column N=lm
            const int kb  = hg * 8;                 // lanes 0-15: K 0..7 then 16..23 ; 16-31: K 8..15 then 24..31
            v8bf h0 = *(const v8bf*)&Bhi[col * LDB + kb];
            v8bf h1 = *(const v8bf*)&Bhi[col * LDB + kb + 16];
            bh[ni] = CAT16(h0, h1);
            v8bf l0 = *(const v8bf*)&Blo[col * LDB + kb];
            v8bf l1 = *(const v8bf*)&Blo[col * LDB + kb + 16];
            bl[ni] = CAT16(l0, l1);
        }

        // ---- per-mi A fragment + split-bf16 triple product ----
#pragma unroll
        for (int mi = 0; mi < 4; ++mi) {
            const int row = waveM + mi * 16 + lm;   // A: lane holds row M=lm
            // lanes 0-15: K 0..7 then 16..23 ; lanes 16-31: K 8..15 then 24..31
            v8bf h0 = *(const v8bf*)&Ahi[row * LDA + hg * 8];
            v8bf h1 = *(const v8bf*)&Ahi[row * LDA + 16 + hg * 8];
            v16bf ah = CAT16(h0, h1);
            v8bf l0 = *(const v8bf*)&Alo[row * LDA + hg * 8];
            v8bf l1 = *(const v8bf*)&Alo[row * LDA + 16 + hg * 8];
            v16bf al = CAT16(l0, l1);
#pragma unroll
            for (int ni = 0; ni < 4; ++ni) {
                acc[mi][ni] = __builtin_amdgcn_wmma_f32_16x16x32_bf16(
                    false, ah, false, bh[ni], (short)0, acc[mi][ni], false, false);
                acc[mi][ni] = __builtin_amdgcn_wmma_f32_16x16x32_bf16(
                    false, ah, false, bl[ni], (short)0, acc[mi][ni], false, false);
                acc[mi][ni] = __builtin_amdgcn_wmma_f32_16x16x32_bf16(
                    false, al, false, bh[ni], (short)0, acc[mi][ni], false, false);
            }
        }

        // ---- convert + store prefetched panel into the other buffer ----
        if (more) store_lds(1 - cur);
        __syncthreads();   // one barrier per iteration
    }

    // ---- store C: VGPR r -> (M = r + 8*hg, N = lm) per 16x16 f32 C/D layout ----
#pragma unroll
    for (int mi = 0; mi < 4; ++mi) {
        const int rowbase = m0 + waveM + mi * 16 + hg * 8;
#pragma unroll
        for (int ni = 0; ni < 4; ++ni) {
            const int col = n0 + waveN + ni * 16 + lm;
#pragma unroll
            for (int r = 0; r < 8; ++r) {
                ob[(size_t)(rowbase + r) * T_DEC + col] = acc[mi][ni][r];
            }
        }
    }
}

extern "C" void kernel_launch(void* const* d_in, const int* in_sizes, int n_in,
                              void* d_out, int out_size, void* d_ws, size_t ws_size,
                              hipStream_t stream) {
    const float* x    = (const float*)d_in[0];
    const float* path = (const float*)d_in[1];
    float*       out  = (float*)d_out;

    const int batch = in_sizes[0] / (DDIM * T_ENC);   // 16

    dim3 grid(T_DEC / BN, DDIM / BM, batch);
    dim3 block(256, 1, 1);
    bmm_wmma_bf16x3<<<grid, block, 0, stream>>>(x, path, out);
}
